// VectorQuantizer_32091995636098
// MI455X (gfx1250) — matrix operand order repaired and checked by reading, not compiled, not run on hardware
//
#include <hip/hip_runtime.h>

#define NROWS  32768
#define DDIM   256
#define NCODES 8192
#define DECAYF 0.99f
#define EPSV   1e-5f

typedef __attribute__((ext_vector_type(16))) __bf16 v16bf;
typedef __attribute__((ext_vector_type(8)))  __bf16 v8bf;
typedef __attribute__((ext_vector_type(8)))  float  v8f;
typedef __attribute__((ext_vector_type(4)))  int    v4i;

#if defined(__has_builtin)
#if __has_builtin(__builtin_amdgcn_global_load_async_to_lds_b128) && \
    __has_builtin(__builtin_amdgcn_s_wait_asynccnt)
#define HAVE_ASYNC 1
#endif
#endif
#ifndef HAVE_ASYNC
#define HAVE_ASYNC 0
#endif

// Builtin signature: void(v4i AS(1)* src, v4i AS(3)* dst, imm offset, imm cpol)
typedef __attribute__((address_space(1))) v4i as1_v4i;
typedef __attribute__((address_space(3))) v4i as3_v4i;
// AS3 pointers are 32-bit; a generic pointer to LDS truncates to the LDS offset
// (flat aperture rule), so integer round-trips are valid addrspace casts here.
#define TO_LDS(p) ((as3_v4i*)(unsigned int)(size_t)(p))
#define TO_GBL(p) ((as1_v4i*)(size_t)(p))

template <int N>
static __device__ __forceinline__ void wait_async_le() {
#if HAVE_ASYNC
  __builtin_amdgcn_s_wait_asynccnt(N);
#endif
}

static __device__ __forceinline__ v8f wmma_bf16(v16bf a, v16bf b, v8f c) {
  // D = A(16x32 bf16) * B(32x16 bf16) + C(16x16 f32)
  return __builtin_amdgcn_wmma_f32_16x16x32_bf16(false, a, false, b, (short)0, c,
                                                 false, false);
}

// ---- K0: split z into bf16 hi/lo -----------------------------------------
__global__ __launch_bounds__(256) void k_split_z(const float* __restrict__ z,
                                                 __bf16* __restrict__ z_hi,
                                                 __bf16* __restrict__ z_lo) {
  size_t i = (size_t)blockIdx.x * 256 + threadIdx.x;
  float x = z[i];
  __bf16 h = (__bf16)x;
  z_hi[i] = h;
  z_lo[i] = (__bf16)(x - (float)h);
}

// ---- K1: split codebook + row norms --------------------------------------
__global__ __launch_bounds__(256) void k_split_codebook(const float* __restrict__ cw,
                                                        __bf16* __restrict__ c_hi,
                                                        __bf16* __restrict__ c_lo,
                                                        float* __restrict__ c_norms) {
  __shared__ float sred[8];
  int row = blockIdx.x, t = threadIdx.x;
  size_t i = (size_t)row * DDIM + t;
  float x = cw[i];
  __bf16 h = (__bf16)x;
  c_hi[i] = h;
  c_lo[i] = (__bf16)(x - (float)h);
  float s = x * x;
#pragma unroll
  for (int m = 16; m >= 1; m >>= 1) s += __shfl_xor(s, m, 32);
  if ((t & 31) == 0) sred[t >> 5] = s;
  __syncthreads();
  if (t == 0) {
    float tot = 0.f;
#pragma unroll
    for (int w = 0; w < 8; ++w) tot += sred[w];
    c_norms[row] = tot;
  }
}

// ---- K2: bf16x3 WMMA distance + fused running argmin ---------------------
// 256 threads (8 waves). Wave w owns rows m0..m0+15; A fragments (hi+lo,
// K=256) live in registers for the whole N loop. Codebook tiles (32 codes x
// 256 dims, hi+lo) are DOUBLE-BUFFERED in LDS and staged with async
// global->LDS copies (ASYNCcnt) so tile fetch overlaps the WMMA chain.
#define TILE_ELEMS (32 * 264)  // padded rows: 264 bf16 (528B = 33*16B) -> conflict-free b128

static __device__ __forceinline__ void stage_tile(int tid, int n0,
                                                  const __bf16* __restrict__ c_hi,
                                                  const __bf16* __restrict__ c_lo,
                                                  __bf16* dst_h, __bf16* dst_l) {
  // 2048 chunks of 16B (32 rows x 512B x {hi,lo}); 8 per thread -> 8 async ops/wave
#pragma unroll
  for (int j = 0; j < 8; ++j) {
    int c   = tid + j * 256;
    int mat = c >> 10;          // 0 = hi, 1 = lo
    int row = (c >> 5) & 31;
    int o16 = c & 31;
    const __bf16* src = (mat ? c_lo : c_hi) + (size_t)(n0 + row) * DDIM + o16 * 8;
    __bf16* dst = (mat ? dst_l : dst_h) + row * 264 + o16 * 8;
#if HAVE_ASYNC
    __builtin_amdgcn_global_load_async_to_lds_b128(TO_GBL(src), TO_LDS(dst), 0, 0);
#else
    *(v8bf*)dst = *(const v8bf*)src;
#endif
  }
}

__global__ __launch_bounds__(256) void k_vq_argmin(
    const __bf16* __restrict__ z_hi, const __bf16* __restrict__ z_lo,
    const __bf16* __restrict__ c_hi, const __bf16* __restrict__ c_lo,
    const float* __restrict__ c_norms,
    int* __restrict__ idx_out, float* __restrict__ idx_f_out) {
  extern __shared__ __bf16 smem[];  // 2 buffers x {hi,lo} x TILE_ELEMS

  const int tid  = threadIdx.x;
  const int lane = tid & 31;
  const int wave = tid >> 5;
  const int hid  = lane >> 4;   // 0: lanes 0-15, 1: lanes 16-31
  const int l16  = lane & 15;
  const int m0   = (blockIdx.x * 8 + wave) * 16;
  const int mrow = m0 + l16;

  // ---- A fragments (ISA 16-bit A layout: lanes<16 hold K {0..7,16..23},
  // lanes>=16 hold K {8..15,24..31} of each 32-wide K step)
  v16bf a_hi[8], a_lo[8];
#pragma unroll
  for (int kk = 0; kk < 8; ++kk) {
    const int kb = kk * 32 + hid * 8;
    const __bf16* ph = z_hi + (size_t)mrow * DDIM + kb;
    const __bf16* pl = z_lo + (size_t)mrow * DDIM + kb;
    v8bf h0 = *(const v8bf*)(ph);
    v8bf h1 = *(const v8bf*)(ph + 16);
    v8bf q0 = *(const v8bf*)(pl);
    v8bf q1 = *(const v8bf*)(pl + 16);
#pragma unroll
    for (int i = 0; i < 8; ++i) {
      a_hi[kk][i] = h0[i]; a_hi[kk][8 + i] = h1[i];
      a_lo[kk][i] = q0[i]; a_lo[kk][8 + i] = q1[i];
    }
  }

  float best_d[8];
  int   best_n[8];
#pragma unroll
  for (int r = 0; r < 8; ++r) { best_d[r] = 3.4e38f; best_n[r] = 0; }

  // prologue: stage tile 0 into buffer 0
  stage_tile(tid, 0, c_hi, c_lo, smem, smem + TILE_ELEMS);

#pragma unroll 1
  for (int it = 0; it < NCODES / 32; ++it) {
    const int n0  = it * 32;
    const int buf = it & 1;
    __bf16* cur_h = smem + buf * (2 * TILE_ELEMS);
    __bf16* cur_l = cur_h + TILE_ELEMS;

    // issue async stage of the NEXT tile into the other buffer
    if (it + 1 < NCODES / 32) {
      __bf16* nxt_h = smem + (buf ^ 1) * (2 * TILE_ELEMS);
      stage_tile(tid, n0 + 32, c_hi, c_lo, nxt_h, nxt_h + TILE_ELEMS);
      wait_async_le<8>();   // 8 newest ops may stay in flight; current tile done
    } else {
      wait_async_le<0>();
    }
    __syncthreads();

    v8f acc0 = {}; v8f acc1 = {};
#pragma unroll
    for (int kk = 0; kk < 8; ++kk) {
      const int ko = kk * 32 + hid * 8;  // lanes<16: K {0..7,16..23}, lanes>=16: K {8..15,24..31}
      const __bf16* bh0 = cur_h + (l16)*264 + ko;
      const __bf16* bl0 = cur_l + (l16)*264 + ko;
      const __bf16* bh1 = cur_h + (16 + l16) * 264 + ko;
      const __bf16* bl1 = cur_l + (16 + l16) * 264 + ko;
      v16bf Bh0, Bl0, Bh1, Bl1;
      v8bf x0, x1;
      x0 = *(const v8bf*)bh0; x1 = *(const v8bf*)(bh0 + 16);
#pragma unroll
      for (int i = 0; i < 8; ++i) { Bh0[i] = x0[i]; Bh0[8 + i] = x1[i]; }
      x0 = *(const v8bf*)bl0; x1 = *(const v8bf*)(bl0 + 16);
#pragma unroll
      for (int i = 0; i < 8; ++i) { Bl0[i] = x0[i]; Bl0[8 + i] = x1[i]; }
      x0 = *(const v8bf*)bh1; x1 = *(const v8bf*)(bh1 + 16);
#pragma unroll
      for (int i = 0; i < 8; ++i) { Bh1[i] = x0[i]; Bh1[8 + i] = x1[i]; }
      x0 = *(const v8bf*)bl1; x1 = *(const v8bf*)(bl1 + 16);
#pragma unroll
      for (int i = 0; i < 8; ++i) { Bl1[i] = x0[i]; Bl1[8 + i] = x1[i]; }

      // bf16x3: hi*hi + lo*hi + hi*lo, f32 accumulate
      acc0 = wmma_bf16(a_hi[kk], Bh0, acc0);
      acc0 = wmma_bf16(a_lo[kk], Bh0, acc0);
      acc0 = wmma_bf16(a_hi[kk], Bl0, acc0);
      acc1 = wmma_bf16(a_hi[kk], Bh1, acc1);
      acc1 = wmma_bf16(a_lo[kk], Bh1, acc1);
      acc1 = wmma_bf16(a_hi[kk], Bl1, acc1);
    }
    __syncthreads();  // tile consumed; next iteration may overwrite this buffer

    // dist = ||c||^2 - 2*(z.c)  (||z||^2 is row-constant -> irrelevant to argmin)
#pragma unroll
    for (int sub = 0; sub < 2; ++sub) {
      int   n  = n0 + sub * 16 + l16;
      float cn = c_norms[n];
#pragma unroll
      for (int r = 0; r < 8; ++r) {
        float a = (sub ? acc1[r] : acc0[r]);
        float d = cn - 2.0f * a;
        if (d < best_d[r]) { best_d[r] = d; best_n[r] = n; }  // n increasing -> first-min ties
      }
    }
  }

  // cross-lane argmin within each 16-lane half (C layout: VGPR r, lanes<16 -> M=r,
  // lanes>=16 -> M=r+8; each lane covered n = l16 (mod 16))
#pragma unroll
  for (int r = 0; r < 8; ++r) {
    float d = best_d[r]; int n = best_n[r];
#pragma unroll
    for (int m = 8; m >= 1; m >>= 1) {
      float od = __shfl_xor(d, m, 32);
      int   on = __shfl_xor(n, m, 32);
      if (od < d || (od == d && on < n)) { d = od; n = on; }
    }
    best_d[r] = d; best_n[r] = n;
  }
  if (l16 == 0) {
#pragma unroll
    for (int r = 0; r < 8; ++r) {
      int m = m0 + r + hid * 8;
      idx_out[m]   = best_n[r];
      idx_f_out[m] = (float)best_n[r];
    }
  }
}

// ---- K3: gather z_q, commitment-loss partials, segment-sum scatter -------
__global__ __launch_bounds__(256) void k_gather_scatter(
    const float* __restrict__ z, const float* __restrict__ cw,
    const int* __restrict__ idx, float* __restrict__ zq_out,
    float* __restrict__ wsum, float* __restrict__ counts,
    float* __restrict__ loss_acc) {
  __shared__ float sred[8];
  const int t = threadIdx.x;
  float part = 0.f;
#pragma unroll 1
  for (int rr = 0; rr < 16; ++rr) {
    int m    = blockIdx.x * 16 + rr;
    int code = idx[m];
    float zv = z[(size_t)m * DDIM + t];
    float q  = cw[(size_t)code * DDIM + t];
    zq_out[(size_t)m * DDIM + t] = q;     // straight-through output == gathered code
    float diff = zv - q;
    part += diff * diff;
    atomicAdd(&wsum[(size_t)code * DDIM + t], zv);
    if (t == 0) atomicAdd(&counts[code], 1.0f);
  }
#pragma unroll
  for (int m = 16; m >= 1; m >>= 1) part += __shfl_xor(part, m, 32);
  if ((t & 31) == 0) sred[t >> 5] = part;
  __syncthreads();
  if (t == 0) {
    float s = 0.f;
#pragma unroll
    for (int w = 0; w < 8; ++w) s += sred[w];
    atomicAdd(loss_acc, s);
  }
}

// ---- K4a: EMA count + global n reduction ---------------------------------
__global__ __launch_bounds__(256) void k_ema_count(const float* __restrict__ ema_count,
                                                   const float* __restrict__ counts,
                                                   float* __restrict__ out_count,
                                                   float* __restrict__ n_acc) {
  __shared__ float sred[8];
  int c = blockIdx.x * 256 + threadIdx.x;
  float nc = DECAYF * ema_count[c] + (1.0f - DECAYF) * counts[c];
  out_count[c] = nc;
  float s = nc;
#pragma unroll
  for (int m = 16; m >= 1; m >>= 1) s += __shfl_xor(s, m, 32);
  if ((threadIdx.x & 31) == 0) sred[threadIdx.x >> 5] = s;
  __syncthreads();
  if (threadIdx.x == 0) {
    float tot = 0.f;
#pragma unroll
    for (int w = 0; w < 8; ++w) tot += sred[w];
    atomicAdd(n_acc, tot);
  }
}

// ---- K4b: EMA weight -----------------------------------------------------
__global__ __launch_bounds__(256) void k_ema_weight(const float* __restrict__ ema_weight,
                                                    const float* __restrict__ wsum,
                                                    float* __restrict__ out_ew) {
  size_t i = (size_t)blockIdx.x * 256 + threadIdx.x;
  out_ew[i] = DECAYF * ema_weight[i] + (1.0f - DECAYF) * wsum[i];
}

// ---- K5: normalized codebook + loss scalar -------------------------------
__global__ __launch_bounds__(256) void k_codebook(const float* __restrict__ out_count,
                                                  const float* __restrict__ out_ew,
                                                  const float* __restrict__ n_acc,
                                                  const float* __restrict__ loss_acc,
                                                  float* __restrict__ out_cb,
                                                  float* __restrict__ out_loss) {
  size_t i = (size_t)blockIdx.x * 256 + threadIdx.x;
  int code = (int)(i >> 8);
  float n  = *n_acc;
  float nc = out_count[code];
  float cs = (nc + EPSV) / (n + (float)NCODES * EPSV) * n;
  out_cb[i] = out_ew[i] / cs;
  if (i == 0) *out_loss = 0.25f * (*loss_acc) / (float)((size_t)NROWS * DDIM);
}

extern "C" void kernel_launch(void* const* d_in, const int* in_sizes, int n_in,
                              void* d_out, int out_size, void* d_ws, size_t ws_size,
                              hipStream_t stream) {
  const float* z          = (const float*)d_in[0];   // 32768 x 256
  const float* cw         = (const float*)d_in[1];   // 8192 x 256
  const float* ema_count  = (const float*)d_in[2];   // 8192
  const float* ema_weight = (const float*)d_in[3];   // 8192 x 256

  // ---- workspace layout (bytes) ----
  char* ws = (char*)d_ws;
  size_t off = 0;
  auto take = [&](size_t bytes) { size_t o = off; off = (off + bytes + 255) & ~(size_t)255; return o; };
  size_t o_zhi  = take((size_t)NROWS * DDIM * 2);
  size_t o_zlo  = take((size_t)NROWS * DDIM * 2);
  size_t o_chi  = take((size_t)NCODES * DDIM * 2);
  size_t o_clo  = take((size_t)NCODES * DDIM * 2);
  size_t o_cn   = take((size_t)NCODES * 4);
  size_t o_cnt  = take((size_t)NCODES * 4);
  size_t o_wsum = take((size_t)NCODES * DDIM * 4);
  size_t o_scl  = take(256);                  // [0]=loss_acc [1]=n_acc
  size_t o_idx  = take((size_t)NROWS * 4);

  __bf16* z_hi = (__bf16*)(ws + o_zhi);
  __bf16* z_lo = (__bf16*)(ws + o_zlo);
  __bf16* c_hi = (__bf16*)(ws + o_chi);
  __bf16* c_lo = (__bf16*)(ws + o_clo);
  float* c_norms  = (float*)(ws + o_cn);
  float* counts   = (float*)(ws + o_cnt);
  float* wsum     = (float*)(ws + o_wsum);
  float* loss_acc = (float*)(ws + o_scl);
  float* n_acc    = (float*)(ws + o_scl) + 1;
  int*   idx      = (int*)(ws + o_idx);

  // ---- d_out layout (f32 elements, reference return order) ----
  float* out = (float*)d_out;
  float* out_zq   = out;                                     // 8388608
  float* out_idx  = out + (size_t)NROWS * DDIM;              // 32768
  float* out_loss = out_idx + NROWS;                         // 1
  float* out_cb   = out_loss + 1;                            // 2097152
  float* out_cnt  = out_cb + (size_t)NCODES * DDIM;          // 8192
  float* out_ew   = out_cnt + NCODES;                        // 2097152

  (void)hipMemsetAsync(counts, 0, (size_t)NCODES * 4, stream);
  (void)hipMemsetAsync(wsum, 0, (size_t)NCODES * DDIM * 4, stream);
  (void)hipMemsetAsync(ws + o_scl, 0, 256, stream);

  k_split_z<<<(NROWS * DDIM) / 256, 256, 0, stream>>>(z, z_hi, z_lo);
  k_split_codebook<<<NCODES, 256, 0, stream>>>(cw, c_hi, c_lo, c_norms);

  // 256 blocks x 8 waves x 16 rows = 32768 rows; dynamic LDS: 2 x {hi,lo} tiles
  const size_t lds_bytes = (size_t)4 * TILE_ELEMS * sizeof(__bf16);  // 67584
  k_vq_argmin<<<NROWS / 128, 256, lds_bytes, stream>>>(z_hi, z_lo, c_hi, c_lo,
                                                       c_norms, idx, out_idx);

  k_gather_scatter<<<NROWS / 16, 256, 0, stream>>>(z, cw, idx, out_zq, wsum,
                                                   counts, loss_acc);

  k_ema_count<<<NCODES / 256, 256, 0, stream>>>(ema_count, counts, out_cnt, n_acc);
  k_ema_weight<<<(NCODES * DDIM) / 256, 256, 0, stream>>>(ema_weight, wsum, out_ew);
  k_codebook<<<(NCODES * DDIM) / 256, 256, 0, stream>>>(out_cnt, out_ew, n_acc,
                                                        loss_acc, out_cb, out_loss);
}
